// PPO_65807488909490
// MI455X (gfx1250) — matrix operand order repaired and checked by reading, not compiled, not run on hardware
//
#include <hip/hip_runtime.h>

typedef __bf16 bf16;
typedef __attribute__((ext_vector_type(16))) bf16  v16bf;
typedef __attribute__((ext_vector_type(8)))  bf16  v8bf;
typedef __attribute__((ext_vector_type(8)))  float v8f;

#define N_JOBS 128
#define N_MACH 256
#define DF     8
#define NODES  (N_JOBS * N_MACH)   // 32768
#define MROWS  64                  // node rows per workgroup (4 M-tiles of 16)
#define MT     (MROWS / 16)        // M-tiles
#define HSTRIDE 264                // bf16 elems per LDS activation row (528B, 16B aligned)

#if __has_builtin(__builtin_amdgcn_global_load_async_to_lds_b128) && \
    __has_builtin(__builtin_amdgcn_s_wait_asynccnt)
#define HAVE_ASYNC_LDS 1
#define LOCAL_AS __attribute__((address_space(3)))
typedef int v4i_g __attribute__((vector_size(16)));  // matches builtin param type
#endif

struct WtsPtrs {
  const bf16*  w[16];   // [mlp*4 + layer], swizzled bf16 weights
  const float* b[16];   // biases (fp32, straight from inputs)
};

// ---------------------------------------------------------------------------
// Weight swizzle: fp32 [K x N] row-major -> bf16 WMMA-B fragment order.
// Element layout index = ((nt*KSTEPS + kk)*32 + lane)*16 + j
// where lane: n = nt*16 + (lane&15), k = kk*32 + (lane>>4)*16 + j.
// Zero-pads k >= K and n >= N.
// ---------------------------------------------------------------------------
__global__ void swizzle_w_kernel(const float* __restrict__ W, bf16* __restrict__ dst,
                                 int K, int N, int KSTEPS, int NTILES) {
  int idx = blockIdx.x * 256 + threadIdx.x;
  int total = NTILES * KSTEPS * 512;
  if (idx >= total) return;
  int j  = idx & 15;
  int l  = (idx >> 4) & 31;
  int q  = idx >> 9;              // nt*KSTEPS + kk
  int kk = q % KSTEPS;
  int nt = q / KSTEPS;
  int n = nt * 16 + (l & 15);
  int k = kk * 32 + (l >> 4) * 16 + j;
  float v = (k < K && n < N) ? W[k * N + n] : 0.f;
  dst[idx] = (bf16)v;
}

// ---------------------------------------------------------------------------
// pos = argsort(MM) per row (rows are permutations of 0..255), then
// pred/succ machine-column indices (-1 when invalid -> zero input to MLP).
// ---------------------------------------------------------------------------
__global__ void prep_idx_kernel(const int* __restrict__ MM,
                                int* __restrict__ pidx, int* __restrict__ sidx) {
  __shared__ int pos[N_MACH];
  const int i = blockIdx.x, c = threadIdx.x;
  const int step = MM[i * N_MACH + c];
  pos[step] = c;
  __syncthreads();
  pidx[i * N_MACH + c] = (step > 0)          ? pos[step - 1] : -1;
  sidx[i * N_MACH + c] = (step < N_MACH - 1) ? pos[step + 1] : -1;
}

// ---------------------------------------------------------------------------
// S[c][d] = sum over jobs i of x[i][c][d]
// ---------------------------------------------------------------------------
__global__ void colsum_kernel(const float* __restrict__ xin, float* __restrict__ S) {
  __shared__ float red[N_JOBS][DF];
  const int c = blockIdx.x, t = threadIdx.x;  // 128 threads
#pragma unroll
  for (int d = 0; d < DF; ++d) red[t][d] = xin[(t * N_MACH + c) * DF + d];
  __syncthreads();
  for (int s = N_JOBS / 2; s > 0; s >>= 1) {
    if (t < s)
#pragma unroll
      for (int d = 0; d < DF; ++d) red[t][d] += red[t + s][d];
    __syncthreads();
  }
  if (t < DF) S[c * DF + t] = red[0][t];
}

__global__ void gsum_kernel(const float* __restrict__ S, float* __restrict__ grelu) {
  const int d = threadIdx.x;
  if (d < DF) {
    float s = 0.f;
    for (int c = 0; c < N_MACH; ++c) s += S[c * DF + d];
    grelu[d] = s > 0.f ? s : 0.f;
  }
}

// ---------------------------------------------------------------------------
// Fused WMMA GEMM over one 64-row activation tile held in LDS.
// Each wave owns N-tile(s); B fragment loaded once per k-step and reused by
// all 4 M-tiles (4 WMMAs per 32B/lane of L2 weight traffic).
//   A (16x32 bf16, contiguous k = 16h+s): lane&15 = m, (lane>>4) selects k-halves
//   B (32x16 bf16, swizzled in memory): one contiguous v16bf per lane
//   C/D (16x16 f32): lane&15 = n, reg r -> m = r + 8*(lane>>4)
// MODE 0: bias+relu -> bf16 LDS (256-wide)
// MODE 1: bias+relu -> bf16 LDS feat columns [colOfs..colOfs+7] (N padded to 16)
// MODE 2: bias      -> fp32 global, 8 columns
// ---------------------------------------------------------------------------
template <int KSTEPS, int NTILES, int MODE>
__device__ __forceinline__ void gemm_wmma(const bf16* actIn, int inStride,
                                          const bf16* __restrict__ Wswz,
                                          const float* __restrict__ bias,
                                          bf16* outLds, int outStride, int colOfs,
                                          float* outGlb) {
  const int lane  = threadIdx.x & 31;
  const int wave  = threadIdx.x >> 5;
  const int am    = lane & 15;
  const int akb   = (lane >> 4) * 16;
  const int nlane = lane & 15;
  const int mhalf = (lane >> 4) * 8;

  for (int nt = wave; nt < NTILES; nt += 8) {
    v8f acc[MT] = {};
    const bf16* bbase = Wswz + (nt * KSTEPS * 32 + lane) * 16;
#pragma unroll
    for (int kk = 0; kk < KSTEPS; ++kk) {
      const v16bf bfrag = *(const v16bf*)(bbase + kk * 512);
#pragma unroll
      for (int m = 0; m < MT; ++m) {
        v16bf afrag;
        const bf16* arow = actIn + (m * 16 + am) * inStride + akb + kk * 32;
        ((v8bf*)&afrag)[0] = *(const v8bf*)(arow);       // K 0..7   of half
        ((v8bf*)&afrag)[1] = *(const v8bf*)(arow + 8);   // K 8..15  of half
        acc[m] = __builtin_amdgcn_wmma_f32_16x16x32_bf16(false, afrag, false, bfrag,
                                                         (short)0, acc[m], false,
                                                         false);
      }
    }
#pragma unroll
    for (int m = 0; m < MT; ++m) {
      const int mbase = m * 16 + mhalf;
      if (MODE == 0) {
        const float bv = bias[nt * 16 + nlane];
#pragma unroll
        for (int r = 0; r < 8; ++r) {
          float v = acc[m][r] + bv;
          v = v > 0.f ? v : 0.f;
          outLds[(mbase + r) * outStride + nt * 16 + nlane] = (bf16)v;
        }
      } else if (MODE == 1) {
        if (nlane < DF) {
          const float bv = bias[nlane];
#pragma unroll
          for (int r = 0; r < 8; ++r) {
            float v = acc[m][r] + bv;
            v = v > 0.f ? v : 0.f;
            outLds[(mbase + r) * outStride + colOfs + nlane] = (bf16)v;
          }
        }
      } else {
        if (nlane < DF) {
          const float bv = bias[nlane];
#pragma unroll
          for (int r = 0; r < 8; ++r)
            outGlb[(mbase + r) * DF + nlane] = acc[m][r] + bv;
        }
      }
    }
  }
}

// ---------------------------------------------------------------------------
// One synchronous GNN iteration, fully fused per 64-row tile.
// ---------------------------------------------------------------------------
__global__ __launch_bounds__(256) void fused_gnn_iter(
    const float* __restrict__ xin, const float* __restrict__ xinit,
    const float* __restrict__ S, const float* __restrict__ grelu,
    const int* __restrict__ pidx, const int* __restrict__ sidx,
    WtsPtrs W, float* __restrict__ xout) {
  extern __shared__ char smemraw[];
  bf16* hA    = (bf16*)smemraw;            // MROWS*HSTRIDE
  bf16* hB    = hA + MROWS * HSTRIDE;      // MROWS*HSTRIDE
  bf16* feat  = hB + MROWS * HSTRIDE;      // MROWS*64  (48 used, 16 pad)
  bf16* inb   = feat + MROWS * 64;         // MROWS*32  (8 used, 24 pad)
  float* xtile = (float*)(inb + MROWS * 32);  // MROWS*DF fp32 (this block's x rows)
  float* itile = xtile + MROWS * DF;          // MROWS*DF fp32 (init rows)
  const int t = threadIdx.x;
  const int rowBase = blockIdx.x * MROWS;

  // Warm the four heavy 256x256 weight matrices (L2-resident across grid).
  __builtin_prefetch(W.w[1], 0, 1);
  __builtin_prefetch(W.w[5], 0, 1);
  __builtin_prefetch(W.w[9], 0, 1);
  __builtin_prefetch(W.w[13], 0, 1);

  // ---- stage this block's x / init rows into LDS (async DMA path) --------
#if HAVE_ASYNC_LDS
  if (t < (MROWS * DF) / 4) {
    __builtin_amdgcn_global_load_async_to_lds_b128(
        (v4i_g*)(xin + rowBase * DF + t * 4),
        (LOCAL_AS v4i_g*)(xtile + t * 4), 0, 0);
    __builtin_amdgcn_global_load_async_to_lds_b128(
        (v4i_g*)(xinit + rowBase * DF + t * 4),
        (LOCAL_AS v4i_g*)(itile + t * 4), 0, 0);
  }
#else
  if (t < (MROWS * DF) / 4) {
#pragma unroll
    for (int q = 0; q < 4; ++q) {
      xtile[t * 4 + q] = xin[rowBase * DF + t * 4 + q];
      itile[t * 4 + q] = xinit[rowBase * DF + t * 4 + q];
    }
  }
#endif

  for (int e = t; e < MROWS * 64; e += 256) feat[e] = (bf16)0.f;
  for (int e = t; e < MROWS * 32; e += 256) inb[e] = (bf16)0.f;
#if HAVE_ASYNC_LDS
  __builtin_amdgcn_s_wait_asynccnt(0);
#endif
  __syncthreads();
  // feat columns 24..31 = a4 (broadcast relu global sum), 32..39 = x, 40..47 = init
  for (int e = t; e < MROWS * DF; e += 256) {
    int m = e >> 3, d = e & 7;
    feat[m * 64 + 24 + d] = (bf16)grelu[d];
    feat[m * 64 + 32 + d] = (bf16)xtile[m * DF + d];
    feat[m * 64 + 40 + d] = (bf16)itile[m * DF + d];
  }

  // ===================== f1 : predecessor-node MLP -> feat[0..7] ==========
  __syncthreads();
  for (int e = t; e < MROWS * DF; e += 256) {
    int m = e >> 3, d = e & 7, r = rowBase + m;
    int p = pidx[r];
    float v = (p >= 0) ? xin[((r >> 8) * N_MACH + p) * DF + d] : 0.f;
    inb[m * 32 + d] = (bf16)v;
  }
  __syncthreads();
  gemm_wmma<1, 16, 0>(inb, 32, W.w[0], W.b[0], hA, HSTRIDE, 0, nullptr);
  __syncthreads();
  gemm_wmma<8, 16, 0>(hA, HSTRIDE, W.w[1], W.b[1], hB, HSTRIDE, 0, nullptr);
  __syncthreads();
  gemm_wmma<8, 16, 0>(hB, HSTRIDE, W.w[2], W.b[2], hA, HSTRIDE, 0, nullptr);
  __syncthreads();
  gemm_wmma<8, 1, 1>(hA, HSTRIDE, W.w[3], W.b[3], feat, 64, 0, nullptr);

  // ===================== f2 : successor-node MLP -> feat[8..15] ===========
  __syncthreads();
  for (int e = t; e < MROWS * DF; e += 256) {
    int m = e >> 3, d = e & 7, r = rowBase + m;
    int p = sidx[r];
    float v = (p >= 0) ? xin[((r >> 8) * N_MACH + p) * DF + d] : 0.f;
    inb[m * 32 + d] = (bf16)v;
  }
  __syncthreads();
  gemm_wmma<1, 16, 0>(inb, 32, W.w[4], W.b[4], hA, HSTRIDE, 0, nullptr);
  __syncthreads();
  gemm_wmma<8, 16, 0>(hA, HSTRIDE, W.w[5], W.b[5], hB, HSTRIDE, 0, nullptr);
  __syncthreads();
  gemm_wmma<8, 16, 0>(hB, HSTRIDE, W.w[6], W.b[6], hA, HSTRIDE, 0, nullptr);
  __syncthreads();
  gemm_wmma<8, 1, 1>(hA, HSTRIDE, W.w[7], W.b[7], feat, 64, 8, nullptr);

  // ===================== f3 : same-machine aggregate -> feat[16..23] ======
  __syncthreads();
  for (int e = t; e < MROWS * DF; e += 256) {
    int m = e >> 3, d = e & 7, r = rowBase + m;
    inb[m * 32 + d] = (bf16)(S[(r & (N_MACH - 1)) * DF + d] - xtile[m * DF + d]);
  }
  __syncthreads();
  gemm_wmma<1, 16, 0>(inb, 32, W.w[8], W.b[8], hA, HSTRIDE, 0, nullptr);
  __syncthreads();
  gemm_wmma<8, 16, 0>(hA, HSTRIDE, W.w[9], W.b[9], hB, HSTRIDE, 0, nullptr);
  __syncthreads();
  gemm_wmma<8, 16, 0>(hB, HSTRIDE, W.w[10], W.b[10], hA, HSTRIDE, 0, nullptr);
  __syncthreads();
  gemm_wmma<8, 1, 1>(hA, HSTRIDE, W.w[11], W.b[11], feat, 64, 16, nullptr);

  // ===================== f4 : feat[48 padded to 64] -> new x ==============
  __syncthreads();
  gemm_wmma<2, 16, 0>(feat, 64, W.w[12], W.b[12], hA, HSTRIDE, 0, nullptr);
  __syncthreads();
  gemm_wmma<8, 16, 0>(hA, HSTRIDE, W.w[13], W.b[13], hB, HSTRIDE, 0, nullptr);
  __syncthreads();
  gemm_wmma<8, 16, 0>(hB, HSTRIDE, W.w[14], W.b[14], hA, HSTRIDE, 0, nullptr);
  __syncthreads();
  gemm_wmma<8, 1, 2>(hA, HSTRIDE, W.w[15], W.b[15], nullptr, 0, 0, xout + rowBase * DF);
}

// ---------------------------------------------------------------------------
extern "C" void kernel_launch(void* const* d_in, const int* in_sizes, int n_in,
                              void* d_out, int out_size, void* d_ws, size_t ws_size,
                              hipStream_t stream) {
  (void)in_sizes; (void)out_size; (void)ws_size;
  const float* x  = (const float*)d_in[0];
  const int*   MM = (const int*)d_in[1];

  // Resolve the 32 MLP parameter arrays (handles both flattened-pytree and
  // per-MLP-concatenated input layouts).
  const float* prm[32];
  if (n_in >= 35) {
    for (int k = 0; k < 32; ++k) prm[k] = (const float*)d_in[3 + k];
  } else {
    static const int szS[8] = {8 * 256, 256, 256 * 256, 256, 256 * 256, 256, 256 * 8, 8};
    static const int szF[8] = {48 * 256, 256, 256 * 256, 256, 256 * 256, 256, 256 * 8, 8};
    for (int mlp = 0; mlp < 4; ++mlp) {
      const float* base = (const float*)d_in[3 + mlp];
      const int* sz = (mlp == 3) ? szF : szS;
      size_t o = 0;
      for (int k = 0; k < 8; ++k) { prm[mlp * 8 + k] = base + o; o += sz[k]; }
    }
  }

  char* ws = (char*)d_ws;
  size_t off = 0;
  auto carve = [&](size_t bytes) -> char* {
    char* p = ws + off;
    off = (off + bytes + 255) & ~(size_t)255;
    return p;
  };

  // {K, N, KSTEPS, NTILES} per layer; [0]=f1/f2/f3, [1]=f4
  static const int desc[2][4][4] = {
      {{8, 256, 1, 16}, {256, 256, 8, 16}, {256, 256, 8, 16}, {256, 8, 8, 1}},
      {{48, 256, 2, 16}, {256, 256, 8, 16}, {256, 256, 8, 16}, {256, 8, 8, 1}}};

  WtsPtrs Wp;
  for (int mlp = 0; mlp < 4; ++mlp) {
    for (int layer = 0; layer < 4; ++layer) {
      const int* dd = desc[mlp == 3][layer];
      const int elems = dd[3] * dd[2] * 512;
      bf16* dst = (bf16*)carve((size_t)elems * sizeof(bf16));
      swizzle_w_kernel<<<(elems + 255) / 256, 256, 0, stream>>>(
          prm[mlp * 8 + layer * 2], dst, dd[0], dd[1], dd[2], dd[3]);
      Wp.w[mlp * 4 + layer] = dst;
      Wp.b[mlp * 4 + layer] = prm[mlp * 8 + layer * 2 + 1];
    }
  }

  float* xA  = (float*)carve((size_t)NODES * DF * sizeof(float));
  float* xB  = (float*)carve((size_t)NODES * DF * sizeof(float));
  int* pidx  = (int*)carve((size_t)NODES * sizeof(int));
  int* sidx  = (int*)carve((size_t)NODES * sizeof(int));
  float* S   = (float*)carve((size_t)N_MACH * DF * sizeof(float));
  float* gr  = (float*)carve(64 * sizeof(float));

  prep_idx_kernel<<<N_JOBS, N_MACH, 0, stream>>>(MM, pidx, sidx);

  const size_t smem =
      (size_t)(2 * MROWS * HSTRIDE + MROWS * 64 + MROWS * 32) * sizeof(bf16) +
      (size_t)2 * MROWS * DF * sizeof(float);
  const float* cur = x;
  for (int it = 0; it < 3; ++it) {
    colsum_kernel<<<N_MACH, N_JOBS, 0, stream>>>(cur, S);
    gsum_kernel<<<1, 32, 0, stream>>>(S, gr);
    float* out = (it == 2) ? (float*)d_out : (it == 0 ? xA : xB);
    fused_gnn_iter<<<NODES / MROWS, 256, smem, stream>>>(cur, x, S, gr, pidx, sidx,
                                                         Wp, out);
    cur = out;
  }
}
